// Conv2dMem_15685220565425
// MI455X (gfx1250) — 8-bit matrix operand order repaired and checked by reading, compiled, not run on hardware
//
#include <hip/hip_runtime.h>


typedef __attribute__((ext_vector_type(8))) int v8i;
typedef __attribute__((ext_vector_type(4))) int v4i;

#define QMAXF 127.0f
#define BN    16
#define CIN   64
#define COUT  128
#define HW    32
#define KDIM  576
#define MROWS 16384

#if __has_builtin(__builtin_amdgcn_global_load_async_to_lds_b128)
#define HAVE_ASYNC_LDS 1
#else
#define HAVE_ASYNC_LDS 0
#endif

__device__ __forceinline__ void async_cp16(void* lds_dst, const void* gsrc) {
#if HAVE_ASYNC_LDS
    __builtin_amdgcn_global_load_async_to_lds_b128(
        (__attribute__((address_space(1))) v4i*)(gsrc),
        (__attribute__((address_space(3))) v4i*)(lds_dst),
         0,  0);
#else
    *(int4*)lds_dst = *(const int4*)gsrc;
#endif
}

__device__ __forceinline__ void wait_async_all() {
#if HAVE_ASYNC_LDS
#if __has_builtin(__builtin_amdgcn_s_wait_asynccnt)
    __builtin_amdgcn_s_wait_asynccnt(0);
#else
    asm volatile("s_wait_asynccnt 0x0" ::: "memory");
#endif
#endif
}

__global__ void init_ws_kernel(unsigned* __restrict__ stats) {
    if (threadIdx.x == 0) { stats[0] = 0u; stats[1] = 0u; }
}

__global__ void absmax_kernel(const float* __restrict__ p, int n,
                              unsigned* __restrict__ out) {
    __shared__ unsigned red[256];
    unsigned best = 0u;
    for (int i = blockIdx.x * blockDim.x + threadIdx.x; i < n;
         i += gridDim.x * blockDim.x) {
        unsigned b = __float_as_uint(fabsf(p[i]));
        if (b > best) best = b;
    }
    red[threadIdx.x] = best;
    __syncthreads();
    for (int s = 128; s > 0; s >>= 1) {
        if (threadIdx.x < s) {
            unsigned o = red[threadIdx.x + s];
            if (o > red[threadIdx.x]) red[threadIdx.x] = o;
        }
        __syncthreads();
    }
    if (threadIdx.x == 0) atomicMax(out, red[0]);
}

__device__ __forceinline__ char quantize_one(float v, float m) {
    float inv = (m > 0.0f) ? (QMAXF / m) : 0.0f;
    float q = rintf(v * inv);
    q = fminf(fmaxf(q, -QMAXF), QMAXF);
    return (char)(int)q;
}

__global__ void quant_w_kernel(const float* __restrict__ w,
                               const unsigned* __restrict__ mw,
                               char* __restrict__ qW, int n) {
    int i = blockIdx.x * blockDim.x + threadIdx.x;
    if (i >= n) return;
    qW[i] = quantize_one(w[i], __uint_as_float(*mw));
}

__global__ void quant_im2col_kernel(const float* __restrict__ x,
                                    const unsigned* __restrict__ mx,
                                    char* __restrict__ qA) {
    int i = blockIdx.x * blockDim.x + threadIdx.x;
    int m = i / KDIM;
    int k = i - m * KDIM;
    int b  = m >> 10;
    int l  = m & 1023;
    int oh = l >> 5, ow = l & 31;
    int c  = k / 9;
    int r  = k - c * 9;
    int kh = r / 3, kw = r - kh * 3;
    int ih = oh + kh - 1, iw = ow + kw - 1;
    float v = 0.0f;
    if (ih >= 0 && ih < HW && iw >= 0 && iw < HW)
        v = x[(((b * CIN + c) << 5) + ih) * HW + iw];
    qA[i] = quantize_one(v, __uint_as_float(*mx));
}

__global__ void __launch_bounds__(256)
gemm_iu8_kernel(const char* __restrict__ qA, const char* __restrict__ qW,
                const unsigned* __restrict__ stats,
                const float* __restrict__ bias, float* __restrict__ out) {
    __shared__ __align__(16) char As[2][128 * 64];
    __shared__ __align__(16) char Bs[2][128 * 64];

    const int tid   = threadIdx.x;
    const int lane  = tid & 31;
    const int wv    = tid >> 5;
    const int half  = lane >> 4;
    const int nn    = lane & 15;
    const int tileM = blockIdx.x * 128;

    v8i acc[8] = {};

    auto stage = [&](int ks, int buf) {
        const long ko = (long)ks * 64;
        #pragma unroll
        for (int rep = 0; rep < 2; ++rep) {
            int i   = tid + rep * 256;
            int row = i >> 2;
            int c16 = (i & 3) << 4;
            async_cp16(&As[buf][row * 64 + c16],
                       &qA[(long)(tileM + row) * KDIM + ko + c16]);
            async_cp16(&Bs[buf][row * 64 + c16],
                       &qW[(long)row * KDIM + ko + c16]);
        }
    };

    stage(0, 0);

    for (int ks = 0; ks < 9; ++ks) {
        const int cur = ks & 1;
        wait_async_all();
        __syncthreads();
        if (ks < 8) stage(ks + 1, cur ^ 1);

        v8i a;
        {
            const char* abase = &As[cur][(wv * 16 + nn) * 64 + half * 16];
            #pragma unroll
            for (int g = 0; g < 4; ++g) {
                int2 t2 = *(const int2*)(abase + (g >> 1) * 32 + (g & 1) * 8);
                a[2 * g]     = t2.x;
                a[2 * g + 1] = t2.y;
            }
        }

        #pragma unroll
        for (int nt = 0; nt < 8; ++nt) {
            const char* bbase = &Bs[cur][(nt * 16 + nn) * 64 + half * 16];
            int4 lo = *(const int4*)(bbase);
            int4 hi = *(const int4*)(bbase + 32);
            v8i bf;
            bf[0] = lo.x; bf[1] = lo.y; bf[2] = lo.z; bf[3] = lo.w;
            bf[4] = hi.x; bf[5] = hi.y; bf[6] = hi.z; bf[7] = hi.w;
            acc[nt] = __builtin_amdgcn_wmma_i32_16x16x64_iu8(
                true, a, true, bf, acc[nt], false, false);
        }
    }

    const float mx = __uint_as_float(stats[0]);
    const float mw = __uint_as_float(stats[1]);
    const float s  = (mx * mw) / (QMAXF * QMAXF);
    const int mbase = tileM + wv * 16 + half * 8;
    #pragma unroll
    for (int nt = 0; nt < 8; ++nt) {
        const int N  = nt * 16 + nn;
        const float bv = bias[N];
        #pragma unroll
        for (int v = 0; v < 8; ++v) {
            int M = mbase + v;
            int b = M >> 10, l = M & 1023;
            out[(((b * COUT) + N) << 10) + l] = s * (float)acc[nt][v] + bv;
        }
    }
}

extern "C" void kernel_launch(void* const* d_in, const int* in_sizes, int n_in,
                              void* d_out, int out_size, void* d_ws, size_t ws_size,
                              hipStream_t stream) {
    const float* x    = (const float*)d_in[0];
    const float* w    = (const float*)d_in[1];
    const float* bias = (const float*)d_in[2];
    float* out        = (float*)d_out;

    unsigned char* ws = (unsigned char*)d_ws;
    unsigned* stats   = (unsigned*)ws;
    char* qA          = (char*)(ws + 64);
    char* qW          = (char*)(ws + 64 + (size_t)MROWS * KDIM);

    init_ws_kernel<<<1, 32, 0, stream>>>(stats);
    absmax_kernel<<<512, 256, 0, stream>>>(x, BN * CIN * HW * HW, stats + 0);
    absmax_kernel<<<128, 256, 0, stream>>>(w, COUT * KDIM, stats + 1);
    quant_im2col_kernel<<<(MROWS * KDIM) / 256, 256, 0, stream>>>(x, stats + 0, qA);
    quant_w_kernel<<<(COUT * KDIM + 255) / 256, 256, 0, stream>>>(w, stats + 1, qW,
                                                                  COUT * KDIM);
    gemm_iu8_kernel<<<MROWS / 128, 256, 0, stream>>>(qA, qW, stats, bias, out);
}
